// KiperwasserDependencyParser_1108101562966
// MI455X (gfx1250) — matrix operand order repaired and checked by reading, not compiled, not run on hardware
//
#include <hip/hip_runtime.h>
#include <hip/hip_bf16.h>
#include <math.h>

// ---------------------------------------------------------------------------
// Kiperwasser dependency parser forward pass for MI455X (gfx1250, wave32).
// GEMMs: v_wmma_f32_16x16x32_bf16, 128x64 block tile, B strip staged in LDS
// by the Tensor Data Mover (tensor_load_to_lds + s_wait_tensorcnt).
// LSTM recurrence: latency-bound VALU. Scorer: tanh-transcendental bound.
// ---------------------------------------------------------------------------

typedef __bf16 bf16;
typedef __attribute__((ext_vector_type(16))) __bf16 v16bf;
typedef __attribute__((ext_vector_type(8)))  __bf16 v8bf;
typedef __attribute__((ext_vector_type(8)))  float  v8f;
typedef __attribute__((ext_vector_type(4)))  unsigned int u32x4;
typedef __attribute__((ext_vector_type(8)))  int i32x8;
typedef __attribute__((ext_vector_type(4)))  int i32x4;

#define SEQ   512
#define WE_D  300
#define PE_D  100
#define IN0   400
#define K0P   416          // IN0 padded to multiple of 32
#define HID   512
#define G4H   2048         // 4*H
#define N0    4096         // both directions' gates side by side
#define IN1   1024
#define MDIM  512

#if defined(__HIP_DEVICE_COMPILE__) && defined(__has_builtin)
# if __has_builtin(__builtin_amdgcn_tensor_load_to_lds)
#  define HAVE_TDM 1
# endif
#endif

__device__ __forceinline__ bf16 f2bf(float f) {
  union { float f; unsigned u; } v; v.f = f;
  unsigned u = v.u;
  unsigned r = u + 0x7FFFu + ((u >> 16) & 1u);   // round-to-nearest-even
  unsigned short h = (unsigned short)(r >> 16);
  return __builtin_bit_cast(bf16, h);
}

__device__ __forceinline__ float sigm(float x) {
  return 1.0f / (1.0f + __expf(-x));
}

// ---------------------------------------------------------------------------
// Embedding gather + concat, emitted directly as padded bf16 (512 x 416).
// ---------------------------------------------------------------------------
__global__ void embed_concat_bf16(const int* __restrict__ wi,
                                  const int* __restrict__ pi,
                                  const float* __restrict__ wemb,
                                  const float* __restrict__ pemb,
                                  bf16* __restrict__ xbf) {
  int idx = blockIdx.x * blockDim.x + threadIdx.x;
  if (idx >= SEQ * K0P) return;
  int s = idx / K0P, k = idx % K0P;
  float v = 0.0f;
  if (k < WE_D)       v = wemb[(size_t)wi[s] * WE_D + k];
  else if (k < IN0)   v = pemb[(size_t)pi[s] * PE_D + (k - WE_D)];
  xbf[idx] = f2bf(v);
}

// f32 [R x C] -> bf16 [R x Cpad] (zero padded)
__global__ void convert_pad_bf16(const float* __restrict__ src,
                                 bf16* __restrict__ dst,
                                 int R, int C, int Cpad) {
  int idx = blockIdx.x * blockDim.x + threadIdx.x;
  if (idx >= R * Cpad) return;
  int r = idx / Cpad, c = idx % Cpad;
  dst[idx] = f2bf(c < C ? src[(size_t)r * C + c] : 0.0f);
}

__global__ void vec_add(const float* __restrict__ a, const float* __restrict__ b,
                        float* __restrict__ dst, int n) {
  int i = blockIdx.x * blockDim.x + threadIdx.x;
  if (i < n) dst[i] = a[i] + b[i];
}

// ---------------------------------------------------------------------------
// C[M x N] = A[M x K] * B[N x K]^T + bias[N]    (bf16 in, f32 out)
// Block tile: 128 rows x 64 cols. 8 waves; wave w owns rows [128*blk+16w, +16)
// and all 4 column sub-tiles (4 accumulators, A fragment reused 4x).
// The 64-row B strip (shared by all 8 waves) is staged into LDS once per
// block by the Tensor Data Mover; waves read B fragments with ds loads.
// Requires M%128==0, N%64==0, K%32==0, K<=65535.
// Fragment layouts: contiguous k order (k = 16*(lane/16) + s) in both A and B (wave32).
// ---------------------------------------------------------------------------
__global__ void __launch_bounds__(256)
wmma_gemm_bt(const bf16* __restrict__ A, int lda,
             const bf16* __restrict__ B, int ldb,
             float* __restrict__ C, int ldc,
             const float* __restrict__ bias,
             int M, int N, int K) {
  extern __shared__ char smem[];
  bf16* Bs = (bf16*)smem;                       // [64][K] row-major
  const int lane = threadIdx.x & 31;
  const int wave = threadIdx.x >> 5;
  const int nStrips = N >> 6;
  const int tn0 = (blockIdx.x % nStrips) << 6;  // 64-col strip base
  const int tm  = ((blockIdx.x / nStrips) << 7) + (wave << 4);

#ifdef HAVE_TDM
  // --- TDM: DMA B[tn0 .. tn0+63][0..K) into LDS (wave 0 issues once) ---
  if (wave == 0) {
    unsigned long long ga = (unsigned long long)(const void*)(B + (size_t)tn0 * ldb);
    unsigned ldsA = (unsigned)(unsigned long long)(void*)Bs;
    u32x4 g0;
    g0[0] = 1u;                                            // count=1 (valid)
    g0[1] = ldsA;                                          // lds_addr
    g0[2] = (unsigned)(ga & 0xFFFFFFFFull);                // global_addr lo
    g0[3] = (unsigned)((ga >> 32) & 0x01FFFFFFull) | (2u << 30); // hi | type=2
    i32x8 g1;
    g1[0] = 1 << 16;                                       // data_size = 2B
    g1[1] = (int)((unsigned)K << 16);                      // tensor_dim0[15:0]
    g1[2] = (int)(64u << 16) | (int)((unsigned)K >> 16);   // tdim1=64 | tdim0 hi
    g1[3] = (int)((unsigned)K << 16);                      // tile_dim0 = K
    g1[4] = 64;                                            // tile_dim1 = 64
    g1[5] = ldb;                                           // tensor_dim0_stride
    g1[6] = 0;
    g1[7] = 0;
    i32x4 z4 = {0, 0, 0, 0};
#if __clang_major__ >= 23
    i32x8 z8 = {0, 0, 0, 0, 0, 0, 0, 0};
    __builtin_amdgcn_tensor_load_to_lds(g0, g1, z4, z4, z8, 0);
#else
    __builtin_amdgcn_tensor_load_to_lds(g0, g1, z4, z4, 0);
#endif
    __builtin_amdgcn_s_wait_tensorcnt(0);
  }
#else
  // Fallback: cooperative global -> LDS copy of the B strip (16B chunks)
  {
    const int c8PerRow = K >> 3;
    for (int idx = threadIdx.x; idx < 64 * c8PerRow; idx += 256) {
      int row = idx / c8PerRow, c8 = idx % c8PerRow;
      *(uint4*)(Bs + (size_t)row * K + c8 * 8) =
          *(const uint4*)(B + (size_t)(tn0 + row) * ldb + c8 * 8);
    }
  }
#endif
  __syncthreads();

  const int half16 = lane & 15;
  // A: lane m in [0,15] holds row m, K {0..15}; lanes 16..31 hold {16..31}
  const bf16* aPtr = A + (size_t)(tm + half16) * lda + ((lane < 16) ? 0 : 16);
  // B fragments from LDS: lane n holds 16 contiguous K of strip row (16j + n)
  const bf16* bPtr = Bs + (size_t)half16 * K + ((lane < 16) ? 0 : 16);

  v8f acc0 = {}, acc1 = {}, acc2 = {}, acc3 = {};
  for (int k0 = 0; k0 < K; k0 += 32) {
    v8bf a0 = *(const v8bf*)(aPtr + k0);
    v8bf a1 = *(const v8bf*)(aPtr + k0 + 8);
    v16bf am;
#pragma unroll
    for (int i = 0; i < 8; ++i) { am[i] = a0[i]; am[i + 8] = a1[i]; }
    if (k0 + 32 < K)
      __builtin_prefetch(aPtr + k0 + 32, 0, 1);   // global_prefetch_b8
    v16bf b0 = *(const v16bf*)(bPtr + k0);
    v16bf b1 = *(const v16bf*)(bPtr + (size_t)16 * K + k0);
    v16bf b2 = *(const v16bf*)(bPtr + (size_t)32 * K + k0);
    v16bf b3 = *(const v16bf*)(bPtr + (size_t)48 * K + k0);
    acc0 = __builtin_amdgcn_wmma_f32_16x16x32_bf16(false, am, false, b0,
                                                   (short)0, acc0, false, false);
    acc1 = __builtin_amdgcn_wmma_f32_16x16x32_bf16(false, am, false, b1,
                                                   (short)0, acc1, false, false);
    acc2 = __builtin_amdgcn_wmma_f32_16x16x32_bf16(false, am, false, b2,
                                                   (short)0, acc2, false, false);
    acc3 = __builtin_amdgcn_wmma_f32_16x16x32_bf16(false, am, false, b3,
                                                   (short)0, acc3, false, false);
  }

  // C layout: VGPR r -> row tm + r (+8 for lanes 16..31), col = sub*16 + (lane&15)
  const int rowOff = (lane < 16) ? 0 : 8;
  float* cRow = C + (size_t)(tm + rowOff) * ldc + tn0 + half16;
#pragma unroll
  for (int sub = 0; sub < 4; ++sub) {
    const v8f acc = (sub == 0) ? acc0 : (sub == 1) ? acc1 : (sub == 2) ? acc2 : acc3;
    const int col = tn0 + sub * 16 + half16;
    const float badd = bias ? bias[col] : 0.0f;
#pragma unroll
    for (int r = 0; r < 8; ++r)
      cRow[(size_t)r * ldc + sub * 16] = acc[r] + badd;
  }
}

// ---------------------------------------------------------------------------
// LSTM scan: one block per direction, 512 threads; thread t owns hidden unit t
// and computes its 4 gate dot products (length 512) against h_{t-1} in LDS.
// xg: [SEQ][4096] (dir*2048 slice).  whh: [2][2048][512].  hout: [SEQ][1024].
// ---------------------------------------------------------------------------
__global__ void __launch_bounds__(512)
lstm_scan(const float* __restrict__ xg,
          const float* __restrict__ whh,
          float* __restrict__ hout) {
  const int dir = blockIdx.x;
  const int t = threadIdx.x;
  __shared__ float hprev[HID];

  const float* w0 = whh + (size_t)dir * G4H * HID + (size_t)t * HID;
  const float* w1 = w0 + (size_t)HID * HID;
  const float* w2 = w0 + (size_t)2 * HID * HID;
  const float* w3 = w0 + (size_t)3 * HID * HID;

  float c = 0.0f;
  hprev[t] = 0.0f;
  __syncthreads();

  for (int step = 0; step < SEQ; ++step) {
    const int s = dir ? (SEQ - 1 - step) : step;
    const float* xgs = xg + (size_t)s * N0 + dir * G4H;
    float gi = xgs[t], gf = xgs[HID + t], gg = xgs[2 * HID + t], go = xgs[3 * HID + t];
#pragma unroll 4
    for (int k = 0; k < HID; ++k) {
      const float hk = hprev[k];
      gi = fmaf(w0[k], hk, gi);
      gf = fmaf(w1[k], hk, gf);
      gg = fmaf(w2[k], hk, gg);
      go = fmaf(w3[k], hk, go);
    }
    gi = sigm(gi); gf = sigm(gf); gg = tanhf(gg); go = sigm(go);
    c = gf * c + gi * gg;
    const float h = go * tanhf(c);
    __syncthreads();
    hprev[t] = h;
    hout[(size_t)s * (2 * HID) + dir * HID + t] = h;
    __syncthreads();
  }
}

// ---------------------------------------------------------------------------
// Scorer + column log-softmax. Block j computes scores[:, j], its logsumexp,
// and the NLL contribution for row heads[j]. b2 cancels in log-softmax.
// ---------------------------------------------------------------------------
__global__ void __launch_bounds__(256)
score_col(const float* __restrict__ Xh, const float* __restrict__ Ym,
          const float* __restrict__ b1, const float* __restrict__ W2,
          const int* __restrict__ heads, float* __restrict__ partial) {
  const int j = blockIdx.x;
  const int tid = threadIdx.x;
  __shared__ float ymb[MDIM];
  __shared__ float w2s[MDIM];
  __shared__ float sc[SEQ];
  __shared__ float red[256];

  for (int m = tid; m < MDIM; m += 256) {
    ymb[m] = Ym[(size_t)j * MDIM + m] + b1[m];
    w2s[m] = W2[m];
  }
  __syncthreads();

  for (int i = tid; i < SEQ; i += 256) {
    const float* xr = Xh + (size_t)i * MDIM;
    float acc = 0.0f;
#pragma unroll 4
    for (int m = 0; m < MDIM; ++m)
      acc = fmaf(tanhf(xr[m] + ymb[m]), w2s[m], acc);
    sc[i] = acc;
  }
  __syncthreads();

  // max over the 512 scores
  red[tid] = fmaxf(sc[tid], sc[tid + 256]);
  __syncthreads();
  for (int off = 128; off > 0; off >>= 1) {
    if (tid < off) red[tid] = fmaxf(red[tid], red[tid + off]);
    __syncthreads();
  }
  const float mx = red[0];
  __syncthreads();

  // sum of exp
  red[tid] = __expf(sc[tid] - mx) + __expf(sc[tid + 256] - mx);
  __syncthreads();
  for (int off = 128; off > 0; off >>= 1) {
    if (tid < off) red[tid] += red[tid + off];
    __syncthreads();
  }
  if (tid == 0)
    partial[j] = sc[heads[j]] - (mx + logf(red[0]));
}

__global__ void __launch_bounds__(256)
loss_reduce(const float* __restrict__ partial, float* __restrict__ out) {
  __shared__ float red[256];
  const int t = threadIdx.x;
  red[t] = partial[t] + partial[t + 256];
  __syncthreads();
  for (int off = 128; off > 0; off >>= 1) {
    if (t < off) red[t] += red[t + off];
    __syncthreads();
  }
  if (t == 0) out[0] = -red[0] / (float)SEQ;
}

// ---------------------------------------------------------------------------
extern "C" void kernel_launch(void* const* d_in, const int* in_sizes, int n_in,
                              void* d_out, int out_size, void* d_ws, size_t ws_size,
                              hipStream_t stream) {
  (void)in_sizes; (void)n_in; (void)out_size; (void)ws_size;
  const int*   word_idx = (const int*)d_in[0];
  const int*   pos_idx  = (const int*)d_in[1];
  const int*   heads    = (const int*)d_in[2];
  const float* word_emb = (const float*)d_in[3];
  const float* pos_emb  = (const float*)d_in[4];
  const float* w_ih0    = (const float*)d_in[5];
  const float* w_hh0    = (const float*)d_in[6];
  const float* b_ih0    = (const float*)d_in[7];
  const float* b_hh0    = (const float*)d_in[8];
  const float* w_ih1    = (const float*)d_in[9];
  const float* w_hh1    = (const float*)d_in[10];
  const float* b_ih1    = (const float*)d_in[11];
  const float* b_hh1    = (const float*)d_in[12];
  const float* W1       = (const float*)d_in[13];
  const float* b1       = (const float*)d_in[14];
  const float* W2       = (const float*)d_in[15];
  float*       out      = (float*)d_out;

  // workspace carve-up (256B aligned)
  char* p = (char*)d_ws;
  auto alloc = [&](size_t bytes) -> void* {
    void* r = (void*)p;
    p += (bytes + 255) & ~(size_t)255;
    return r;
  };
  bf16*  x_bf     = (bf16*) alloc((size_t)SEQ * K0P * 2);
  bf16*  wih0_bf  = (bf16*) alloc((size_t)N0 * K0P * 2);
  bf16*  wih1_bf  = (bf16*) alloc((size_t)N0 * IN1 * 2);
  bf16*  W1_bf    = (bf16*) alloc((size_t)MDIM * 2048 * 2);
  bf16*  h0_bf    = (bf16*) alloc((size_t)SEQ * IN1 * 2);
  bf16*  h1_bf    = (bf16*) alloc((size_t)SEQ * IN1 * 2);
  float* xg0      = (float*)alloc((size_t)SEQ * N0 * 4);
  float* xg1      = (float*)alloc((size_t)SEQ * N0 * 4);
  float* h0       = (float*)alloc((size_t)SEQ * IN1 * 4);
  float* h1       = (float*)alloc((size_t)SEQ * IN1 * 4);
  float* b0c      = (float*)alloc((size_t)N0 * 4);
  float* b1c      = (float*)alloc((size_t)N0 * 4);
  float* Xh       = (float*)alloc((size_t)SEQ * MDIM * 4);
  float* Ym       = (float*)alloc((size_t)SEQ * MDIM * 4);
  float* partial  = (float*)alloc((size_t)SEQ * 4);

  const int T = 256;
  // combined biases
  vec_add<<<(N0 + T - 1) / T, T, 0, stream>>>(b_ih0, b_hh0, b0c, N0);
  vec_add<<<(N0 + T - 1) / T, T, 0, stream>>>(b_ih1, b_hh1, b1c, N0);

  // embeddings -> padded bf16 input
  embed_concat_bf16<<<(SEQ * K0P + T - 1) / T, T, 0, stream>>>(
      word_idx, pos_idx, word_emb, pos_emb, x_bf);

  // layer-0 input GEMM (both directions fused along N)
  convert_pad_bf16<<<((size_t)N0 * K0P + T - 1) / T, T, 0, stream>>>(
      w_ih0, wih0_bf, N0, IN0, K0P);
  wmma_gemm_bt<<<(SEQ / 128) * (N0 / 64), 256, 64 * K0P * 2, stream>>>(
      x_bf, K0P, wih0_bf, K0P, xg0, N0, b0c, SEQ, N0, K0P);
  lstm_scan<<<2, 512, 0, stream>>>(xg0, w_hh0, h0);

  // layer-1 input GEMM
  convert_pad_bf16<<<((size_t)SEQ * IN1 + T - 1) / T, T, 0, stream>>>(
      h0, h0_bf, SEQ, IN1, IN1);
  convert_pad_bf16<<<((size_t)N0 * IN1 + T - 1) / T, T, 0, stream>>>(
      w_ih1, wih1_bf, N0, IN1, IN1);
  wmma_gemm_bt<<<(SEQ / 128) * (N0 / 64), 256, 64 * IN1 * 2, stream>>>(
      h0_bf, IN1, wih1_bf, IN1, xg1, N0, b1c, SEQ, N0, IN1);
  lstm_scan<<<2, 512, 0, stream>>>(xg1, w_hh1, h1);

  // biaffine projections Xh = h1 @ A^T, Ym = h1 @ B^T  (A|B = W1 halves)
  convert_pad_bf16<<<((size_t)SEQ * IN1 + T - 1) / T, T, 0, stream>>>(
      h1, h1_bf, SEQ, IN1, IN1);
  convert_pad_bf16<<<((size_t)MDIM * 2048 + T - 1) / T, T, 0, stream>>>(
      W1, W1_bf, MDIM, 2048, 2048);
  wmma_gemm_bt<<<(SEQ / 128) * (MDIM / 64), 256, 64 * IN1 * 2, stream>>>(
      h1_bf, IN1, W1_bf, 2048, Xh, MDIM, nullptr, SEQ, MDIM, IN1);
  wmma_gemm_bt<<<(SEQ / 128) * (MDIM / 64), 256, 64 * IN1 * 2, stream>>>(
      h1_bf, IN1, W1_bf + IN1, 2048, Ym, MDIM, nullptr, SEQ, MDIM, IN1);

  // scorer + per-column log-softmax + final loss
  score_col<<<SEQ, 256, 0, stream>>>(Xh, Ym, b1, W2, heads, partial);
  loss_reduce<<<1, 256, 0, stream>>>(partial, out);
}
